// GATMergeGraph_38482906972409
// MI455X (gfx1250) — compile-verified, matrix operand order checked by reading, not run on hardware
//
#include <hip/hip_runtime.h>
#include <hip/hip_bf16.h>
#include <math.h>

// ---- problem constants (from reference) ----
#define B_     64
#define S_     24
#define D_     300
#define N_     256
#define E_     1024
#define HL_    512
#define HEADS_ 4
#define HD_    128
#define OUT_   512
#define NA_    3000
#define KP_    320            // D padded to multiple of 32
#define KT_    (KP_ / 32)     // 10 k-tiles
#define G4_    (4 * HL_)      // 2048 LSTM gates

typedef __attribute__((ext_vector_type(16))) __bf16 v16bf;
typedef __attribute__((ext_vector_type(8)))  float  v8f;
typedef __attribute__((ext_vector_type(4)))  int    v4i;
typedef __attribute__((ext_vector_type(8)))  int    v8i;

__device__ inline __bf16 to_bf16(float f) {
  __hip_bfloat16 h = __float2bfloat16(f);
  return *reinterpret_cast<__bf16*>(&h);
}

// CDNA5 async global -> LDS copy, 16B per lane, tracked with ASYNCcnt.
__device__ inline void async_g2l_b128(void* lds, const void* g) {
  unsigned off = (unsigned)(uintptr_t)lds;   // low 32 bits of generic shared ptr = LDS byte offset
  asm volatile("global_load_async_to_lds_b128 %0, %1, off"
               :: "v"(off), "v"(g) : "memory");
}

// ---------------- utility ----------------
__global__ void k_zero(float* p, int n) {
  int i = blockIdx.x * blockDim.x + threadIdx.x;
  if (i < n) p[i] = 0.f;
}

// seq[b,s,d] = word_emb[question[b,s], d]
__global__ void k_embed_q(const int* __restrict__ question,
                          const float* __restrict__ emb,
                          float* __restrict__ seq) {
  int i = blockIdx.x * blockDim.x + threadIdx.x;
  if (i >= B_ * S_ * D_) return;
  int d = i % D_;
  int bs = i / D_;
  seq[i] = emb[(long)question[bs] * D_ + d];
}

// ---------------- packed-fragment producers ----------------
// A-fragment element map (16-bit A 16x32, wave32, ISA 7.12.2):
//   lane l: m = l&15, hi = l>>4
//   element e (0..15): k = kt*32 + 8*hi + e + (e>=8 ? 8 : 0)
// Packed layout: Ap[(mt*KT + kt)*512 + lane*16 + e]  (32B contiguous per lane)
__global__ void k_pack_feat(const int* __restrict__ toks,
                            const float* __restrict__ emb,
                            __bf16* __restrict__ Ap,
                            long mtiles, int tokper) {
  long i = (long)blockIdx.x * blockDim.x + threadIdx.x;
  long total = mtiles * KT_ * 512;
  if (i >= total) return;
  int  rr   = (int)(i & 511);
  long tile = i >> 9;
  int  kt   = (int)(tile % KT_);
  long mt   = tile / KT_;
  int lane = rr >> 4, e = rr & 15;
  int m = lane & 15, hi = lane >> 4;
  int k = kt * 32 + 8 * hi + e + ((e >= 8) ? 8 : 0);
  long row = mt * 16 + m;
  float v = 0.f;
  if (k < D_) {
    const int* t = toks + row * tokper;
    #pragma unroll
    for (int j = 0; j < 4; ++j) v += emb[(long)t[j] * D_ + k];
  }
  Ap[i] = to_bf16(v);
}

// B-fragment element map (16-bit B 32x16 mirrored from A):
//   lane l: n = n0 + (l&15), hi = l>>4; element e: k = kt*32 + 8*hi + e + (e>=8?8:0)
// Packed layout: Bp[(kt*NT + nt)*512 + lane*16 + e]
__global__ void k_pack_w(const float* __restrict__ W,
                         __bf16* __restrict__ Bp,
                         int K, int Nc, int NT) {
  long i = (long)blockIdx.x * blockDim.x + threadIdx.x;
  long total = (long)KT_ * NT * 512;
  if (i >= total) return;
  int  rr   = (int)(i & 511);
  long tile = i >> 9;
  int nt = (int)(tile % NT);
  int kt = (int)(tile / NT);
  int lane = rr >> 4, e = rr & 15;
  int n = nt * 16 + (lane & 15);
  int k = kt * 32 + 8 * (lane >> 4) + e + ((e >= 8) ? 8 : 0);
  float v = (k < K && n < Nc) ? W[(long)k * Nc + n] : 0.f;
  Bp[i] = to_bf16(v);
}

// ---------------- WMMA GEMM with TDM + async LDS staging ----------------
// Block = 4 waves = 64x64 tile. B tiles (shared by all waves) staged once via
// a single TENSOR_LOAD_TO_LDS (D# descriptor, TENSORcnt); each wave's A strip
// staged via GLOBAL_LOAD_ASYNC_TO_LDS_B128 (ASYNCcnt). Compute runs from LDS.
// mode 0: C[row*ldc+col] = acc
// mode 2: relu + per-graph reduce: atomicAdd(edges_sum[(row>>10)*300 + col]) for col<300
__global__ __launch_bounds__(128)
void k_wmma_gemm(const __bf16* __restrict__ Ap,
                 const __bf16* __restrict__ Bp,
                 float* __restrict__ C,
                 int NT, int ldc, int mode,
                 float* __restrict__ edges_sum) {
  __shared__ __bf16 sB[KT_ * 4 * 512];      // 40 KB: 10 k-tiles x 4 n-tiles x 1KB
  __shared__ __bf16 sA[4][KT_ * 512];       // 10 KB per wave
  int lane = threadIdx.x & 31;
  int wave = threadIdx.x >> 5;
  long mt  = (long)blockIdx.y * 4 + wave;
  int  nt0 = blockIdx.x * 4;

  // --- stage B via Tensor Data Mover: 2D tile, 10 rows x 4096B, stride NT*1024B ---
  if (wave == 0) {
    unsigned long long ga = (unsigned long long)(uintptr_t)(Bp + (size_t)nt0 * 512);
    unsigned ldsB = (unsigned)(uintptr_t)&sB[0];
    unsigned td0  = (unsigned)NT * 512;     // tensor_dim0 / dim0 stride (elements)
    v4i g0;
    g0[0] = 1;                                               // count=1, user descriptor
    g0[1] = (int)ldsB;                                       // lds_addr (bytes)
    g0[2] = (int)(unsigned)(ga & 0xffffffffull);             // global_addr[31:0]
    g0[3] = (int)(((unsigned)(ga >> 32) & 0x1ffffffu) | (2u << 30)); // addr[56:32] | type=2
    v8i g1;
    g1[0] = 0x00010000;                                      // wg_mask=0, data_size=2B
    g1[1] = (int)((td0 & 0xffffu) << 16);                    // tensor_dim0[15:0]
    g1[2] = (int)((td0 >> 16) | ((unsigned)KT_ << 16));      // tensor_dim0[31:16] | tensor_dim1[15:0]
    g1[3] = (int)(2048u << 16);                              // tile_dim0 = 2048 elements
    g1[4] = KT_;                                             // tile_dim1 = 10 rows
    g1[5] = (int)td0;                                        // tensor_dim0_stride[31:0]
    g1[6] = 0;                                               // stride hi / dim1 stride lo
    g1[7] = 0;
    asm volatile("tensor_load_to_lds %0, %1" :: "s"(g0), "s"(g1) : "memory");
  }

  // --- stage this wave's A strip (KT*16x32 fragments, 10KB) via async loads ---
  {
    const char* gA = (const char*)(Ap + (mt * KT_) * 512);
    char* lA = (char*)&sA[wave][0];
    for (int i = lane; i < (KT_ * 1024) / 16; i += 32)       // 640 x 16B lane-chunks
      async_g2l_b128(lA + i * 16, gA + i * 16);
  }
  asm volatile("s_wait_asynccnt 0x0" ::: "memory");
  __builtin_amdgcn_s_wait_tensorcnt(0);                      // no-op for waves 1..3
  __syncthreads();

  v8f acc[4] = {};
  for (int kt = 0; kt < KT_; ++kt) {
    v16bf a = *(const v16bf*)(&sA[wave][kt * 512 + lane * 16]);
    #pragma unroll
    for (int j = 0; j < 4; ++j) {
      v16bf b = *(const v16bf*)(&sB[(kt * 4 + j) * 512 + lane * 16]);
      acc[j] = __builtin_amdgcn_wmma_f32_16x16x32_bf16(
          /*neg_a=*/false, a, /*neg_b=*/false, b,
          /*c_mod=*/(short)0, acc[j], /*reuse_a=*/false, /*reuse_b=*/false);
    }
  }

  int hi = lane >> 4, nl = lane & 15;
  long rowbase = mt * 16 + 8 * hi;
  if (mode == 0) {
    #pragma unroll
    for (int j = 0; j < 4; ++j) {
      long col = (long)(nt0 + j) * 16 + nl;
      #pragma unroll
      for (int v = 0; v < 8; ++v)
        C[(rowbase + v) * ldc + col] = acc[j][v];
    }
  } else {
    #pragma unroll
    for (int j = 0; j < 4; ++j) {
      int col = (nt0 + j) * 16 + nl;
      if (col < D_) {
        #pragma unroll
        for (int v = 0; v < 8; ++v) {
          long row = rowbase + v;
          float val = acc[j][v];
          val = val > 0.f ? val : 0.f;                  // relu(edge_out)
          int bb = (int)(row >> 10);                    // row / E
          atomicAdd(&edges_sum[bb * D_ + col], val);    // per-graph edge sum
        }
      }
    }
  }
}

// ---------------- LSTM step (B=64 blocks, 256 threads) ----------------
__global__ void k_lstm_step(const float* __restrict__ seq,
                            const int* __restrict__ qmask,
                            const float* __restrict__ W_ih,
                            const float* __restrict__ W_hh,
                            const float* __restrict__ bl,
                            float* __restrict__ hbuf,
                            float* __restrict__ cbuf, int t) {
  __shared__ float xs[D_];
  __shared__ float hs[HL_];
  __shared__ float gs[G4_];
  int b = blockIdx.x, tid = threadIdx.x;
  for (int i = tid; i < D_; i += 256) xs[i] = seq[((long)b * S_ + t) * D_ + i];
  for (int i = tid; i < HL_; i += 256) hs[i] = hbuf[b * HL_ + i];
  __syncthreads();
  for (int g = tid; g < G4_; g += 256) {
    float acc = bl[g];
    for (int k = 0; k < D_; ++k)  acc += xs[k] * W_ih[(long)k * G4_ + g];
    for (int k = 0; k < HL_; ++k) acc += hs[k] * W_hh[(long)k * G4_ + g];
    gs[g] = acc;
  }
  __syncthreads();
  bool m = qmask[b * S_ + t] > 0;
  for (int u = tid; u < HL_; u += 256) {
    float ig = 1.f / (1.f + __expf(-gs[u]));
    float fg = 1.f / (1.f + __expf(-gs[HL_ + u]));
    float gg = tanhf(gs[2 * HL_ + u]);
    float og = 1.f / (1.f + __expf(-gs[3 * HL_ + u]));
    float cold = cbuf[b * HL_ + u];
    float cnew = fg * cold + ig * gg;
    float hnew = og * tanhf(cnew);
    if (m) { cbuf[b * HL_ + u] = cnew; hbuf[b * HL_ + u] = hnew; }
  }
}

// ---------------- attention ----------------
__device__ inline unsigned f32_to_ordered(float f) {
  unsigned b = __float_as_uint(f);
  return (b & 0x80000000u) ? ~b : (b | 0x80000000u);
}
__device__ inline float ordered_to_f32(unsigned o) {
  unsigned b = (o & 0x80000000u) ? (o & 0x7FFFFFFFu) : ~o;
  return __uint_as_float(b);
}

__global__ void k_logits(const float* __restrict__ h, const float* __restrict__ r,
                         const int* __restrict__ src_ids, const int* __restrict__ dst_ids,
                         const float* __restrict__ a_src, const float* __restrict__ a_dst,
                         const float* __restrict__ a_edge,
                         float* __restrict__ logits, unsigned* __restrict__ mx) {
  int i = blockIdx.x * blockDim.x + threadIdx.x;
  if (i >= B_ * E_ * HEADS_) return;
  int head = i & 3;
  int be = i >> 2;
  int b = be >> 10;                 // E = 1024
  int sg = b * N_ + src_ids[be];
  int dg = b * N_ + dst_ids[be];
  const float* hsv = h + (long)sg * OUT_ + head * HD_;
  const float* hdv = h + (long)dg * OUT_ + head * HD_;
  const float* rv  = r + (long)be * OUT_ + head * HD_;
  const float* as = a_src + head * HD_;
  const float* ad = a_dst + head * HD_;
  const float* ae = a_edge + head * HD_;
  float acc = 0.f;
  for (int k = 0; k < HD_; ++k)
    acc += hsv[k] * as[k] + hdv[k] * ad[k] + rv[k] * ae[k];
  acc = acc > 0.f ? acc : 0.2f * acc;            // leaky_relu(0.2)
  logits[i] = acc;
  atomicMax(&mx[dg * HEADS_ + head], f32_to_ordered(acc));
}

__global__ void k_expden(const float* __restrict__ logits,
                         const int* __restrict__ dst_ids,
                         const unsigned* __restrict__ mx,
                         float* __restrict__ ex, float* __restrict__ den) {
  int i = blockIdx.x * blockDim.x + threadIdx.x;
  if (i >= B_ * E_ * HEADS_) return;
  int head = i & 3;
  int be = i >> 2;
  int b = be >> 10;
  int dg = b * N_ + dst_ids[be];
  float m = ordered_to_f32(mx[dg * HEADS_ + head]);
  float e = __expf(logits[i] - m);
  ex[i] = e;
  atomicAdd(&den[dg * HEADS_ + head], e);
}

__global__ void k_aggr(const float* __restrict__ h, const float* __restrict__ r,
                       const float* __restrict__ ex, const float* __restrict__ den,
                       const int* __restrict__ src_ids, const int* __restrict__ dst_ids,
                       float* __restrict__ node_out) {
  long i = (long)blockIdx.x * blockDim.x + threadIdx.x;
  if (i >= (long)B_ * E_ * OUT_) return;
  int d = (int)(i & 511);
  long be = i >> 9;
  int head = d >> 7;
  int b = (int)(be >> 10);
  int sg = b * N_ + src_ids[be];
  int dg = b * N_ + dst_ids[be];
  float alpha = ex[be * HEADS_ + head] / (den[dg * HEADS_ + head] + 1e-9f);
  float val = alpha * (h[(long)sg * OUT_ + d] + r[(long)be * OUT_ + d]);
  atomicAdd(&node_out[(long)dg * OUT_ + d], val);
}

// nodes_sum[b,d] = sum_n elu(node_out[b,n,d])
__global__ void k_nodes_sum(const float* __restrict__ node_out,
                            float* __restrict__ nodes_sum) {
  int b = blockIdx.x, d = threadIdx.x;   // 512 threads
  float acc = 0.f;
  for (int n = 0; n < N_; ++n) {
    float v = node_out[((long)(b * N_ + n)) * OUT_ + d];
    acc += (v > 0.f) ? v : (__expf(v) - 1.f);
  }
  nodes_sum[b * OUT_ + d] = acc;
}

// ---------------- small fusion GEMMs (M=64, VALU) ----------------
__global__ void k_sgemm(const float* __restrict__ A, const float* __restrict__ W,
                        const float* __restrict__ bias, const float* __restrict__ add,
                        float* __restrict__ C, int M, int K, int N) {
  int i = blockIdx.x * blockDim.x + threadIdx.x;
  if (i >= M * N) return;
  int n = i % N, m = i / N;
  float acc = bias ? bias[n] : 0.f;
  const float* a = A + (long)m * K;
  for (int k = 0; k < K; ++k) acc += a[k] * W[(long)k * N + n];
  if (add) acc += add[i];
  C[i] = acc;
}

__global__ void k_concat(const float* __restrict__ x, const float* __restrict__ y,
                         float* __restrict__ out) {
  int i = blockIdx.x * blockDim.x + threadIdx.x;
  if (i >= B_ * 1024) return;
  int n = i & 1023, m = i >> 10;
  out[i] = (n < 512) ? x[m * 512 + n] : y[m * 512 + (n - 512)];
}

// ---------------- launcher ----------------
extern "C" void kernel_launch(void* const* d_in, const int* in_sizes, int n_in,
                              void* d_out, int out_size, void* d_ws, size_t ws_size,
                              hipStream_t stream) {
  (void)in_sizes; (void)n_in; (void)out_size; (void)ws_size;
  const int*   question    = (const int*)d_in[0];
  const int*   qmask       = (const int*)d_in[1];
  const int*   node_tokens = (const int*)d_in[2];
  const int*   edge_tokens = (const int*)d_in[3];
  const int*   src_ids     = (const int*)d_in[4];
  const int*   dst_ids     = (const int*)d_in[5];
  const float* word_emb    = (const float*)d_in[6];
  const float* W_ih        = (const float*)d_in[7];
  const float* W_hh        = (const float*)d_in[8];
  const float* b_lstm      = (const float*)d_in[9];
  const float* W_node      = (const float*)d_in[10];
  const float* W_rel       = (const float*)d_in[11];
  const float* a_src       = (const float*)d_in[12];
  const float* a_dst       = (const float*)d_in[13];
  const float* a_edge      = (const float*)d_in[14];
  const float* W_edge_upd  = (const float*)d_in[15];
  const float* W_qnode     = (const float*)d_in[16];
  const float* b_qnode     = (const float*)d_in[17];
  const float* W_qedge     = (const float*)d_in[18];
  const float* b_qedge     = (const float*)d_in[19];
  const float* W_edge_prj  = (const float*)d_in[20];
  const float* b_edge_prj  = (const float*)d_in[21];
  const float* W_mid       = (const float*)d_in[22];
  const float* b_mid       = (const float*)d_in[23];
  const float* W_pre       = (const float*)d_in[24];
  const float* b_pre       = (const float*)d_in[25];

  char* wsb = (char*)d_ws;
  size_t off = 0;
  auto alloc = [&](size_t bytes) -> char* {
    char* p = wsb + off;
    off = (off + bytes + 255) & ~(size_t)255;
    return p;
  };

  const long BN = (long)B_ * N_;      // 16384
  const long BE = (long)B_ * E_;      // 65536

  float*  seq       = (float*)alloc((size_t)B_ * S_ * D_ * 4);
  float*  hbuf      = (float*)alloc((size_t)B_ * HL_ * 4);
  float*  cbuf      = (float*)alloc((size_t)B_ * HL_ * 4);
  __bf16* Apn       = (__bf16*)alloc((size_t)BN * KP_ * 2);
  __bf16* Ape       = (__bf16*)alloc((size_t)BE * KP_ * 2);
  __bf16* Bpn       = (__bf16*)alloc((size_t)KP_ * 512 * 2);
  __bf16* Bpr       = (__bf16*)alloc((size_t)KP_ * 512 * 2);
  __bf16* Bpu       = (__bf16*)alloc((size_t)KP_ * KP_ * 2);
  float*  h         = (float*)alloc((size_t)BN * OUT_ * 4);
  float*  r         = (float*)alloc((size_t)BE * OUT_ * 4);
  float*  logits    = (float*)alloc((size_t)BE * HEADS_ * 4);
  float*  ex        = (float*)alloc((size_t)BE * HEADS_ * 4);
  unsigned* mx      = (unsigned*)alloc((size_t)BN * HEADS_ * 4);
  float*  den       = (float*)alloc((size_t)BN * HEADS_ * 4);
  float*  node_out  = (float*)alloc((size_t)BN * OUT_ * 4);
  float*  nodes_sum = (float*)alloc((size_t)B_ * OUT_ * 4);
  float*  edges_sum = (float*)alloc((size_t)B_ * D_ * 4);
  float*  qe        = (float*)alloc((size_t)B_ * D_ * 4);
  float*  qn        = (float*)alloc((size_t)B_ * OUT_ * 4);
  float*  qedges    = (float*)alloc((size_t)B_ * OUT_ * 4);
  float*  qcat      = (float*)alloc((size_t)B_ * 1024 * 4);
  float*  midb      = (float*)alloc((size_t)B_ * 1536 * 4);

  auto zero = [&](float* p, long n) {
    k_zero<<<(int)((n + 255) / 256), 256, 0, stream>>>(p, (int)n);
  };
  zero(hbuf, B_ * HL_);
  zero(cbuf, B_ * HL_);
  zero((float*)mx, BN * HEADS_);       // ordered-uint 0 == -inf sentinel
  zero(den, BN * HEADS_);
  zero(node_out, BN * OUT_);
  zero(nodes_sum, B_ * OUT_);
  zero(edges_sum, B_ * D_);

  // embeddings + packed features + packed weights
  k_embed_q<<<(B_ * S_ * D_ + 255) / 256, 256, 0, stream>>>(question, word_emb, seq);
  {
    long tot = (BN / 16) * KT_ * 512;
    k_pack_feat<<<(int)((tot + 255) / 256), 256, 0, stream>>>(node_tokens, word_emb, Apn, BN / 16, 4);
  }
  {
    long tot = (BE / 16) * KT_ * 512;
    k_pack_feat<<<(int)((tot + 255) / 256), 256, 0, stream>>>(edge_tokens, word_emb, Ape, BE / 16, 4);
  }
  {
    long tot = (long)KT_ * 32 * 512;
    k_pack_w<<<(int)((tot + 255) / 256), 256, 0, stream>>>(W_node, Bpn, D_, 512, 32);
    k_pack_w<<<(int)((tot + 255) / 256), 256, 0, stream>>>(W_rel, Bpr, D_, 512, 32);
  }
  {
    long tot = (long)KT_ * 20 * 512;
    k_pack_w<<<(int)((tot + 255) / 256), 256, 0, stream>>>(W_edge_upd, Bpu, D_, D_, 20);
  }

  // LSTM scan (sequential steps on stream)
  for (int t = 0; t < S_; ++t)
    k_lstm_step<<<B_, 256, 0, stream>>>(seq, qmask, W_ih, W_hh, b_lstm, hbuf, cbuf, t);

  // big GEMMs via WMMA (B staged by TDM, A by async global->LDS)
  {
    dim3 g(32 / 4, (unsigned)(BN / 16 / 4));      // h = nf @ W_node  [16384x512]
    k_wmma_gemm<<<g, 128, 0, stream>>>(Apn, Bpn, h, 32, 512, 0, nullptr);
  }
  {
    dim3 g(32 / 4, (unsigned)(BE / 16 / 4));      // r = ef @ W_rel   [65536x512]
    k_wmma_gemm<<<g, 128, 0, stream>>>(Ape, Bpr, r, 32, 512, 0, nullptr);
  }
  {
    dim3 g(20 / 4, (unsigned)(BE / 16 / 4));      // edges_sum += relu(ef @ W_edge_upd)
    k_wmma_gemm<<<g, 128, 0, stream>>>(Ape, Bpu, nullptr, 20, 0, 2, edges_sum);
  }

  // GAT attention
  {
    int tot = (int)(BE * HEADS_);
    k_logits<<<(tot + 255) / 256, 256, 0, stream>>>(h, r, src_ids, dst_ids,
                                                    a_src, a_dst, a_edge, logits, mx);
    k_expden<<<(tot + 255) / 256, 256, 0, stream>>>(logits, dst_ids, mx, ex, den);
  }
  {
    long tot = BE * OUT_;
    k_aggr<<<(int)((tot + 255) / 256), 256, 0, stream>>>(h, r, ex, den, src_ids, dst_ids, node_out);
  }
  k_nodes_sum<<<B_, 512, 0, stream>>>(node_out, nodes_sum);

  // fusion MLP (q_emb == hbuf after scan)
  k_sgemm<<<(B_ * D_ + 255) / 256, 256, 0, stream>>>(hbuf, W_qedge, b_qedge, edges_sum,
                                                     qe, B_, HL_, D_);
  k_sgemm<<<(B_ * OUT_ + 255) / 256, 256, 0, stream>>>(hbuf, W_qnode, b_qnode, nodes_sum,
                                                       qn, B_, HL_, OUT_);
  k_sgemm<<<(B_ * OUT_ + 255) / 256, 256, 0, stream>>>(qe, W_edge_prj, b_edge_prj, nullptr,
                                                       qedges, B_, D_, OUT_);
  k_concat<<<(B_ * 1024 + 255) / 256, 256, 0, stream>>>(qedges, qn, qcat);
  k_sgemm<<<(B_ * 1536 + 255) / 256, 256, 0, stream>>>(qcat, W_mid, b_mid, nullptr,
                                                       midb, B_, 1024, 1536);
  k_sgemm<<<(B_ * NA_ + 255) / 256, 256, 0, stream>>>(midb, W_pre, b_pre, nullptr,
                                                      (float*)d_out, B_, 1536, NA_);
}
